// GlobalAttentionReadout_82952998355882
// MI455X (gfx1250) — matrix operand order repaired and checked by reading, not compiled, not run on hardware
//
#include <hip/hip_runtime.h>
#include <hip/hip_bf16.h>

#define N_NODES 100000
#define HDIM    512
#define GSEG    512

static_assert(N_NODES % 32 == 0, "tile logic assumes 32-row tiles divide N");

typedef __attribute__((ext_vector_type(16))) _Float16 v16h;
typedef __attribute__((ext_vector_type(8)))  _Float16 v8h;
typedef __attribute__((ext_vector_type(4)))  _Float16 v4h;
typedef __attribute__((ext_vector_type(8)))  float    v8f;
typedef __attribute__((ext_vector_type(4)))  float    v4f;
typedef __attribute__((ext_vector_type(4)))  unsigned u32x4;
typedef __attribute__((ext_vector_type(8)))  int      i32x8;
typedef __attribute__((ext_vector_type(4)))  int      i32x4;

// ---------- fragment helpers ----------
// A-matrix 16x32 f16 (ISA 7.12.2): per lane, K = off+0..7 and K = off+16..23
static __device__ inline v16h frag_split(const _Float16* p) {
  union { v16h v; v8h h[2]; } u;
  u.h[0] = *(const v8h*)(p);
  u.h[1] = *(const v8h*)(p + 16);
  return u.v;
}
// B-matrix 32x16 f16: per lane (column), K = off+0..7 and K = off+16..23
static __device__ inline v16h frag_contig(const _Float16* p) {
  union { v16h v; v8h h[2]; } u;
  u.h[0] = *(const v8h*)(p);
  u.h[1] = *(const v8h*)(p + 16);
  return u.v;
}

static __device__ inline unsigned f2ord(float f) {
  unsigned u = __float_as_uint(f);
  return (u & 0x80000000u) ? ~u : (u | 0x80000000u);
}
static __device__ inline float ord2f(unsigned u) {
  unsigned v = (u & 0x80000000u) ? (u & 0x7fffffffu) : ~u;
  return __uint_as_float(v);
}

// ---------- TDM: load 32x512 f32 tile of x into LDS (one issue per wave) ----------
static __device__ inline void tdm_load_x_tile(const float* gsrc, unsigned lds_addr) {
  unsigned long long ga = (unsigned long long)(size_t)gsrc;
  u32x4 g0;
  g0[0] = 1u;                                   // count=1, user D#
  g0[1] = lds_addr;                             // lds_addr (bytes)
  g0[2] = (unsigned)(ga & 0xffffffffu);         // global_addr[31:0]
  g0[3] = (unsigned)((ga >> 32) & 0x01ffffffu)  // global_addr[56:32]
        | 0x80000000u;                          // type=2 ("image") in bits[127:126]
  i32x8 g1;
  g1[0] = 0x00020000;                           // wg_mask=0, data_size=2 (4B)
  g1[1] = (int)(512u << 16);                    // tensor_dim0 = 512
  g1[2] = (int)(32u << 16);                     // tensor_dim1 = 32
  g1[3] = (int)(512u << 16);                    // tile_dim0 = 512
  g1[4] = 32;                                   // tile_dim1 = 32, tile_dim2 = 0
  g1[5] = 512;                                  // tensor_dim0_stride = 512
  g1[6] = 0;
  g1[7] = 0;
  i32x4 gz = {0, 0, 0, 0};
#if __clang_major__ >= 23
  i32x8 gz8 = {0, 0, 0, 0, 0, 0, 0, 0};
  __builtin_amdgcn_tensor_load_to_lds(g0, g1, gz, gz, gz8, 0);
#else
  __builtin_amdgcn_tensor_load_to_lds(g0, g1, gz, gz, 0);
#endif
}

// ---------- async global->LDS stage of one 16KB weight tile (issue only) ----------
static __device__ inline void stage_w_nowait(const _Float16* src, unsigned lds_base, int t) {
  #pragma unroll
  for (int j = 0; j < 16; ++j) {
    int idx = t + j * 64;                       // 1024 x 16B chunks, 64 threads
    asm volatile("global_load_async_to_lds_b128 %0, %1, off"
                 :: "v"(lds_base + idx * 16),
                    "v"((const void*)((const char*)src + idx * 16))
                 : "memory");
  }
}
static __device__ inline void wait_async0() {
  asm volatile("s_wait_asynccnt 0x0" ::: "memory");
}

// ---------- one-time f32 -> f16 conversion of the x tile (in LDS) ----------
static __device__ inline void convert_tile(const float* xs32, _Float16* xh, int t) {
  #pragma unroll 8
  for (int j = 0; j < 64; ++j) {
    int i = t + j * 64;                         // 4096 x v4 chunks
    v4f a = ((const v4f*)xs32)[i];
    v4h h;
    #pragma unroll
    for (int q = 0; q < 4; ++q) h[q] = (_Float16)a[q];
    ((v4h*)xh)[i] = h;
  }
}

// ---------- pipelined column-tile GEMM with fused row-vector epilogue ----------
// sacc[i] += relu( (x @ W)[row_i, col] + bias[col] ) * wvec[col],  summed over cols
template <int CT>
static __device__ inline void score_gemm(const _Float16* __restrict__ wT,
                                         const float* __restrict__ bias,
                                         const float* __restrict__ wvec,
                                         const _Float16* arow, _Float16* wbuf,
                                         unsigned wlds, int t, int lmod, int boff,
                                         float* sacc) {
  stage_w_nowait(wT, wlds, t);
  wait_async0();
  __syncthreads();
  for (int ct = 0; ct < CT; ++ct) {
    const _Float16* curw = wbuf + (ct & 1) * 16 * HDIM;
    if (ct + 1 < CT)
      stage_w_nowait(wT + (size_t)(ct + 1) * 16 * HDIM,
                     wlds + ((ct + 1) & 1) * 16 * HDIM * 2, t);
    v8f c = {};
    const _Float16* bcol = curw + lmod * HDIM + boff;
    #pragma unroll
    for (int ks = 0; ks < 16; ++ks) {
      v16h a = frag_split(arow + ks * 32);
      v16h b = frag_contig(bcol + ks * 32);
      c = __builtin_amdgcn_wmma_f32_16x16x32_f16(false, a, false, b, (short)0, c, false, false);
    }
    const float bv = bias[ct * 16 + lmod];
    const float wv = wvec[ct * 16 + lmod];
    #pragma unroll
    for (int i = 0; i < 8; ++i) {
      float v = c[i] + bv;
      v = v > 0.0f ? v : 0.0f;
      sacc[i] += v * wv;
    }
    wait_async0();        // next tile staged
    __syncthreads();      // all waves done with curw; buffer reusable
  }
}

// ---------- prep kernels ----------
__global__ void init_kernel(float* emb, unsigned* segmax, float* segsum) {
  int i = blockIdx.x * 256 + threadIdx.x;
  if (i < GSEG * HDIM) emb[i] = 0.0f;
  if (i < 2 * GSEG) { segmax[i] = 0u; segsum[i] = 0.0f; }
}

// WT[n*512 + k] = (f16) W[k*cols + n]   (K is always HDIM=512)
__global__ void convw_kernel(const float* __restrict__ W, _Float16* __restrict__ WT, int cols) {
  int i = blockIdx.x * 256 + threadIdx.x;
  if (i >= HDIM * cols) return;
  int k = i / cols, n = i % cols;
  WT[n * HDIM + k] = (_Float16)W[i];
}

__global__ void wbar_kernel(const float* __restrict__ gw2, float* __restrict__ wbar) {
  int k = blockIdx.x * 256 + threadIdx.x;
  if (k >= HDIM) return;
  float s = 0.0f;
  #pragma unroll
  for (int h = 0; h < 8; ++h) s += gw2[k * 8 + h];
  wbar[k] = s * 0.125f;
}

// ---------- WMMA kernel 1: both score vectors ----------
// s1[n] = relu(x@W1g + b1) @ w2g      (gate pre-softmax)
// s2[n] = relu(x@Wg1 + bg1) @ wbar    (mean-head score pre-softmax)
__global__ __launch_bounds__(64) void scores_kernel(
    const float* __restrict__ x,
    const _Float16* __restrict__ w1gT, const float* __restrict__ b1g,
    const float* __restrict__ w2g,
    const _Float16* __restrict__ wg1T, const float* __restrict__ bg1,
    const float* __restrict__ wbar,
    float* __restrict__ s1, float* __restrict__ s2) {
  __shared__ float    xs32[32 * HDIM];       // 64 KB raw f32 x-tile (TDM)
  __shared__ _Float16 xh[32 * HDIM];         // 32 KB converted f16 x-tile
  __shared__ _Float16 wbuf[2 * 16 * HDIM];   // 2 x 16 KB weight double buffer
  const int row0 = blockIdx.x * 32;
  const int t = threadIdx.x;
  const int wave = t >> 5, lane = t & 31;

  if (wave == 0) {
    tdm_load_x_tile(x + (size_t)row0 * HDIM, (unsigned)(size_t)xs32);
    __builtin_amdgcn_s_wait_tensorcnt((short)0);
  }
  __syncthreads();
  convert_tile(xs32, xh, t);
  __syncthreads();

  const int lmod = lane & 15;
  const int aoff = (lane >> 4) << 3;   // 0 or 8
  const int boff = (lane >> 4) << 3;   // 0 or 8
  const _Float16* arow = &xh[(wave * 16 + lmod) * HDIM + aoff];
  const unsigned wlds = (unsigned)(size_t)wbuf;

  float sacc[8];
  #pragma unroll
  for (int i = 0; i < 8; ++i) sacc[i] = 0.0f;
  score_gemm<16>(w1gT, b1g, w2g, arow, wbuf, wlds, t, lmod, boff, sacc);
  #pragma unroll
  for (int i = 0; i < 8; ++i) {
    #pragma unroll
    for (int m = 1; m < 16; m <<= 1) sacc[i] += __shfl_xor(sacc[i], m, 32);
  }
  if (lmod == 0) {
    int rbase = row0 + wave * 16 + aoff;
    #pragma unroll
    for (int i = 0; i < 8; ++i) s1[rbase + i] = sacc[i];
  }

  #pragma unroll
  for (int i = 0; i < 8; ++i) sacc[i] = 0.0f;
  score_gemm<32>(wg1T, bg1, wbar, arow, wbuf, wlds, t, lmod, boff, sacc);
  #pragma unroll
  for (int i = 0; i < 8; ++i) {
    #pragma unroll
    for (int m = 1; m < 16; m <<= 1) sacc[i] += __shfl_xor(sacc[i], m, 32);
  }
  if (lmod == 0) {
    int rbase = row0 + wave * 16 + aoff;
    #pragma unroll
    for (int i = 0; i < 8; ++i) s2[rbase + i] = sacc[i];
  }
}

// ---------- segment softmax helper kernels ----------
__global__ void segmax_kernel(const float* __restrict__ s1, const float* __restrict__ s2,
                              const long long* __restrict__ batch, unsigned* __restrict__ segmax) {
  int i = blockIdx.x * 256 + threadIdx.x;
  if (i >= 2 * N_NODES) return;
  int n = (i < N_NODES) ? i : i - N_NODES;
  int b = (i < N_NODES) ? 0 : 1;
  float v = b ? s2[n] : s1[n];
  atomicMax(&segmax[b * GSEG + (int)batch[n]], f2ord(v));
}

__global__ void expsum_kernel(float* __restrict__ s1, float* __restrict__ s2,
                              const long long* __restrict__ batch,
                              const unsigned* __restrict__ segmax, float* __restrict__ segsum) {
  int i = blockIdx.x * 256 + threadIdx.x;
  if (i >= 2 * N_NODES) return;
  int n = (i < N_NODES) ? i : i - N_NODES;
  int b = (i < N_NODES) ? 0 : 1;
  int seg = (int)batch[n];
  float* s = b ? s2 : s1;
  float e = __expf(s[n] - ord2f(segmax[b * GSEG + seg]));
  s[n] = e;                                // in-place: score -> exp
  atomicAdd(&segsum[b * GSEG + seg], e);
}

__global__ void final_kernel(float* __restrict__ s1, const float* __restrict__ s2,
                             const long long* __restrict__ batch,
                             const float* __restrict__ segsum, float* __restrict__ attn) {
  int i = blockIdx.x * 256 + threadIdx.x;
  if (i >= 2 * N_NODES) return;
  int n = (i < N_NODES) ? i : i - N_NODES;
  int seg = (int)batch[n];
  if (i < N_NODES) {
    s1[n] = s1[n] / (segsum[seg] + 1e-16f);            // gate (kept in ws)
  } else {
    attn[n] = s2[n] / (segsum[GSEG + seg] + 1e-16f);   // output attention
  }
}

// ---------- WMMA kernel 2: gated segment-sum embedding ----------
__global__ __launch_bounds__(64) void embed_kernel(
    const float* __restrict__ x, const _Float16* __restrict__ wnT,
    const float* __restrict__ bn, const float* __restrict__ gate,
    const long long* __restrict__ batch, float* __restrict__ emb) {
  __shared__ float    xs32[32 * HDIM];
  __shared__ _Float16 xh[32 * HDIM];
  __shared__ _Float16 wbuf[2 * 16 * HDIM];
  const int row0 = blockIdx.x * 32;
  const int t = threadIdx.x;
  const int wave = t >> 5, lane = t & 31;

  if (wave == 0) {
    tdm_load_x_tile(x + (size_t)row0 * HDIM, (unsigned)(size_t)xs32);
    __builtin_amdgcn_s_wait_tensorcnt((short)0);
  }
  __syncthreads();
  convert_tile(xs32, xh, t);
  __syncthreads();

  const int lmod = lane & 15;
  const int aoff = (lane >> 4) << 3;
  const int boff = (lane >> 4) << 3;
  const _Float16* arow = &xh[(wave * 16 + lmod) * HDIM + aoff];
  const unsigned wlds = (unsigned)(size_t)wbuf;

  float ga[8]; int sg[8];
  const int rbase = row0 + wave * 16 + aoff;
  #pragma unroll
  for (int i = 0; i < 8; ++i) {
    ga[i] = gate[rbase + i];
    sg[i] = (int)batch[rbase + i];
  }

  stage_w_nowait(wnT, wlds, t);
  wait_async0();
  __syncthreads();
  for (int ct = 0; ct < 32; ++ct) {
    const _Float16* curw = wbuf + (ct & 1) * 16 * HDIM;
    if (ct + 1 < 32)
      stage_w_nowait(wnT + (size_t)(ct + 1) * 16 * HDIM,
                     wlds + ((ct + 1) & 1) * 16 * HDIM * 2, t);
    v8f c = {};
    const _Float16* bcol = curw + lmod * HDIM + boff;
    #pragma unroll
    for (int ks = 0; ks < 16; ++ks) {
      v16h a = frag_split(arow + ks * 32);
      v16h b = frag_contig(bcol + ks * 32);
      c = __builtin_amdgcn_wmma_f32_16x16x32_f16(false, a, false, b, (short)0, c, false, false);
    }
    const int col = ct * 16 + lmod;
    const float bv = bn[col];
    // batch is sorted: run-length combine per-segment before the f32 atomic
    float acc = 0.0f; int cur = sg[0];
    #pragma unroll
    for (int i = 0; i < 8; ++i) {
      float v = c[i] + bv;
      v = v > 0.0f ? v : 0.0f;
      v *= ga[i];
      if (sg[i] != cur) {
        atomicAdd(&emb[cur * HDIM + col], acc);
        acc = 0.0f; cur = sg[i];
      }
      acc += v;
    }
    atomicAdd(&emb[cur * HDIM + col], acc);
    wait_async0();
    __syncthreads();
  }
}

// ---------- workspace layout (bytes) ----------
static constexpr size_t OFF_W1GT = 0;                                   // 256*512 f16
static constexpr size_t OFF_WNT  = OFF_W1GT + (size_t)256 * HDIM * 2;   // 512*512 f16
static constexpr size_t OFF_WG1T = OFF_WNT  + (size_t)HDIM * HDIM * 2;  // 512*512 f16
static constexpr size_t OFF_WBAR = OFF_WG1T + (size_t)HDIM * HDIM * 2;  // 512 f32
static constexpr size_t OFF_S1   = OFF_WBAR + (size_t)HDIM * 4;         // N f32
static constexpr size_t OFF_S2   = OFF_S1   + (size_t)N_NODES * 4;      // N f32
static constexpr size_t OFF_SMAX = OFF_S2   + (size_t)N_NODES * 4;      // 2G u32
static constexpr size_t OFF_SSUM = OFF_SMAX + (size_t)2 * GSEG * 4;     // 2G f32

extern "C" void kernel_launch(void* const* d_in, const int* in_sizes, int n_in,
                              void* d_out, int out_size, void* d_ws, size_t ws_size,
                              hipStream_t stream) {
  (void)in_sizes; (void)n_in; (void)out_size; (void)ws_size;
  const float*     x     = (const float*)d_in[0];
  const long long* batch = (const long long*)d_in[1];
  const float*     w1g   = (const float*)d_in[2];
  const float*     b1g   = (const float*)d_in[3];
  const float*     w2g   = (const float*)d_in[4];
  // d_in[5] ga_g_b2: scalar, cancels exactly under segment softmax
  const float*     wn    = (const float*)d_in[6];
  const float*     bn    = (const float*)d_in[7];
  const float*     wg1   = (const float*)d_in[8];
  const float*     bg1   = (const float*)d_in[9];
  const float*     gw2   = (const float*)d_in[10];
  // d_in[11] g_b2: mean over heads is constant per node, cancels under softmax

  char* ws = (char*)d_ws;
  _Float16* w1gT  = (_Float16*)(ws + OFF_W1GT);
  _Float16* wnT   = (_Float16*)(ws + OFF_WNT);
  _Float16* wg1T  = (_Float16*)(ws + OFF_WG1T);
  float*    wbar  = (float*)(ws + OFF_WBAR);
  float*    s1    = (float*)(ws + OFF_S1);
  float*    s2    = (float*)(ws + OFF_S2);
  unsigned* smax  = (unsigned*)(ws + OFF_SMAX);
  float*    ssum  = (float*)(ws + OFF_SSUM);

  float* emb  = (float*)d_out;                  // (G, H)
  float* attn = (float*)d_out + GSEG * HDIM;    // (N,)

  init_kernel<<<(GSEG * HDIM + 255) / 256, 256, 0, stream>>>(emb, smax, ssum);
  convw_kernel<<<(HDIM * 256 + 255) / 256, 256, 0, stream>>>(w1g, w1gT, 256);
  convw_kernel<<<(HDIM * HDIM + 255) / 256, 256, 0, stream>>>(wn, wnT, HDIM);
  convw_kernel<<<(HDIM * HDIM + 255) / 256, 256, 0, stream>>>(wg1, wg1T, HDIM);
  wbar_kernel<<<2, 256, 0, stream>>>(gw2, wbar);

  scores_kernel<<<N_NODES / 32, 64, 0, stream>>>(x, w1gT, b1g, w2g, wg1T, bg1, wbar, s1, s2);

  int tg = (2 * N_NODES + 255) / 256;
  segmax_kernel<<<tg, 256, 0, stream>>>(s1, s2, batch, smax);
  expsum_kernel<<<tg, 256, 0, stream>>>(s1, s2, batch, smax, ssum);
  final_kernel<<<tg, 256, 0, stream>>>(s1, s2, batch, ssum, attn);

  embed_kernel<<<N_NODES / 32, 64, 0, stream>>>(x, wnT, bn, s1, batch, emb);
}
